// MambaBlock_78323023610179
// MI455X (gfx1250) — matrix operand order repaired and checked by reading, not compiled, not run on hardware
//
#include <hip/hip_runtime.h>
#include <hip/hip_bf16.h>
#include <math.h>

// ---------------------------------------------------------------------------
// Mamba block for MI455X (gfx1250, wave32, WMMA + TDM).
// GEMMs: bf16 v_wmma_f32_16x16x32_bf16, f32 accumulate.
// A row-tiles staged to LDS once per block via the Tensor Data Mover.
// ---------------------------------------------------------------------------

#define D_MODEL 1024
#define D_STATE 16
#define D_CONV  4
#define D_INNER 2048
#define DT_RANK 128
#define B_SZ    2
#define SEQ     2048
#define ROWS    (B_SZ * SEQ)            // 4096
#define NPROJ   (DT_RANK + 2 * D_STATE) // 160

typedef __bf16 bf16;
typedef __attribute__((ext_vector_type(16))) __bf16 v16bf;
typedef __attribute__((ext_vector_type(8)))  __bf16 v8bf;
typedef __attribute__((ext_vector_type(8)))  float  v8f;
typedef unsigned uint32x4 __attribute__((ext_vector_type(4)));
typedef unsigned uint32x8 __attribute__((ext_vector_type(8)));

// ---------------- workspace layout (bytes) ----------------
#define ALIGN256(x) (((x) + 255) & ~size_t(255))
static const size_t OFF_WIN   = 0;                                                    // W_in  bf16  4096x1024
static const size_t OFF_WXP   = ALIGN256(OFF_WIN  + size_t(2*D_INNER)*D_MODEL*2);     // W_xproj bf16 160x2048
static const size_t OFF_WDT   = ALIGN256(OFF_WXP  + size_t(NPROJ)*D_INNER*2);         // W_dt  bf16  2048x128
static const size_t OFF_WOUT  = ALIGN256(OFF_WDT  + size_t(D_INNER)*DT_RANK*2);       // W_out bf16  1024x2048
static const size_t OFF_XN    = ALIGN256(OFF_WOUT + size_t(D_MODEL)*D_INNER*2);       // xn    bf16  4096x1024
static const size_t OFF_XZ    = ALIGN256(OFF_XN   + size_t(ROWS)*D_MODEL*2);          // xz    bf16  4096x4096
static const size_t OFF_U     = ALIGN256(OFF_XZ   + size_t(ROWS)*2*D_INNER*2);        // u     bf16  4096x2048
static const size_t OFF_PROJ  = ALIGN256(OFF_U    + size_t(ROWS)*D_INNER*2);          // proj  f32   4096x160
static const size_t OFF_DTLOW = ALIGN256(OFF_PROJ + size_t(ROWS)*NPROJ*4);            // dtlow bf16  4096x128
static const size_t OFF_DELTA = ALIGN256(OFF_DTLOW+ size_t(ROWS)*DT_RANK*2);          // delta f32   4096x2048
static const size_t OFF_YG    = ALIGN256(OFF_DELTA+ size_t(ROWS)*D_INNER*4);          // yg    bf16  4096x2048

// ---------------- fp32 -> bf16 weight conversion ----------------
__global__ void f2bf_kernel(const float* __restrict__ src, bf16* __restrict__ dst, int n) {
    int i = blockIdx.x * blockDim.x + threadIdx.x;
    if (i < n) dst[i] = (bf16)src[i];
}

// ---------------- LayerNorm -> bf16 ----------------
__global__ void ln_kernel(const float* __restrict__ x, const float* __restrict__ g,
                          const float* __restrict__ b, bf16* __restrict__ xn) {
    __shared__ float red[256];
    int row = blockIdx.x;
    const float* xr = x + (size_t)row * D_MODEL;
    int tid = threadIdx.x;

    float s = 0.f;
    for (int i = tid; i < D_MODEL; i += 256) s += xr[i];
    red[tid] = s; __syncthreads();
    for (int off = 128; off > 0; off >>= 1) {
        if (tid < off) red[tid] += red[tid + off];
        __syncthreads();
    }
    float mu = red[0] * (1.f / D_MODEL);
    __syncthreads();

    float v = 0.f;
    for (int i = tid; i < D_MODEL; i += 256) { float d = xr[i] - mu; v += d * d; }
    red[tid] = v; __syncthreads();
    for (int off = 128; off > 0; off >>= 1) {
        if (tid < off) red[tid] += red[tid + off];
        __syncthreads();
    }
    float rstd = rsqrtf(red[0] * (1.f / D_MODEL) + 1e-5f);

    bf16* out = xn + (size_t)row * D_MODEL;
    for (int i = tid; i < D_MODEL; i += 256)
        out[i] = (bf16)((xr[i] - mu) * rstd * g[i] + b[i]);
}

// ---------------- WMMA GEMM with TDM-staged A tile ----------------
//  C[M,N] = A[M,K] * W[N,K]^T.  K % 32 == 0, M,N % 16 == 0.
//  grid = (ceil(N/16/8), M/16), block = 256 (8 waves).
//  Block's 16xK A row-tile is DMA'd to LDS by the Tensor Data Mover once,
//  then all 8 waves (8 consecutive N-tiles) read fragments from LDS.
// EPI 0: store bf16 to outB (stride N)
// EPI 1: store f32 to outF (stride N); cols < DT_RANK duplicated as bf16 to aux
// EPI 2: softplus(acc + bias[col]) -> outF (stride N)
// EPI 3: acc + resid[row*N+col]    -> outF (stride N)
template <int EPI>
__global__ void wmma_gemm_tdm_kernel(const bf16* __restrict__ A, const bf16* __restrict__ W,
                                     int N, int K,
                                     const float* __restrict__ bias,
                                     const float* __restrict__ resid,
                                     float* __restrict__ outF,
                                     bf16* __restrict__ outB,
                                     bf16* __restrict__ aux) {
    extern __shared__ bf16 smem[];                 // 16 x K bf16 A tile
    const int lane   = threadIdx.x & 31;
    const int wave   = threadIdx.x >> 5;
    const int mt     = blockIdx.y;
    const int nt     = blockIdx.x * 8 + wave;
    const int tilesN = N >> 4;

    // ---- TDM: stage A[mt*16 .. mt*16+15, 0..K) into LDS (wave 0 issues) ----
    if (wave == 0) {
        const unsigned ldsBase = __builtin_amdgcn_groupstaticsize(); // dynamic LDS start
        const unsigned long long ga =
            (unsigned long long)(const void*)(A + (size_t)mt * 16 * K);
        const unsigned uK = (unsigned)K;
        uint32x4 g0;
        g0[0] = 1u;                                          // count=1, user mode
        g0[1] = ldsBase;                                     // lds_addr (bytes)
        g0[2] = (unsigned)ga;                                // global_addr[31:0]
        g0[3] = (unsigned)((ga >> 32) & 0x1FFFFFFu) | (2u << 30); // addr[56:32] | type=2
        uint32x8 g1;
        g1[0] = (1u << 16);                  // workgroup_mask=0, data_size=1 (2 bytes)
        g1[1] = (uK & 0xFFFFu) << 16;        // tensor_dim0[15:0]   (bits 63:48)
        g1[2] = ((uK >> 16) & 0xFFFFu) | (16u << 16); // tensor_dim0[31:16] | tensor_dim1[15:0]
        g1[3] = (uK & 0xFFFFu) << 16;        // tensor_dim1[31:16]=0 | tile_dim0 (bits 127:112)
        g1[4] = 16u;                         // tile_dim1=16, tile_dim2=0
        g1[5] = uK;                          // tensor_dim0_stride[31:0]
        g1[6] = 0u;                          // stride0[47:32]=0, tensor_dim1_stride[15:0]=0
        g1[7] = 0u;                          // tensor_dim1_stride[47:16]=0
        asm volatile("tensor_load_to_lds %0, %1" :: "s"(g0), "s"(g1) : "memory");
        __builtin_amdgcn_s_wait_tensorcnt(0);
    }
    __syncthreads();

    if (nt < tilesN) {                         // wave-uniform guard: EXEC stays all-ones
        const int r    = lane & 15;
        const int half = lane >> 4;

        const bf16* arow = smem + (size_t)r * K;            // A fragment row (LDS)
        const bf16* brow = W + (size_t)(nt * 16 + r) * K;   // B fragment: col = lane%16

        v8f acc = {};
        for (int k = 0; k < K; k += 32) {
            // A 16x32 (bf16): lanes 0-15 hold K 0..7 | 16..23, lanes 16-31: K 8..15 | 24..31
            v8bf a0 = *(const v8bf*)(arow + k + half * 8);
            v8bf a1 = *(const v8bf*)(arow + k + 16 + half * 8);
            // B 32x16 (bf16): lanes 0-15 hold K 0..7 | 16..23, lanes 16-31: K 8..15 | 24..31
            v8bf b0 = *(const v8bf*)(brow + k + half * 8);
            v8bf b1 = *(const v8bf*)(brow + k + 16 + half * 8);
            v16bf av, bv;
#pragma unroll
            for (int i = 0; i < 8; ++i) { av[i] = a0[i]; av[8 + i] = a1[i]; }
#pragma unroll
            for (int i = 0; i < 8; ++i) { bv[i] = b0[i]; bv[8 + i] = b1[i]; }
            if (k + 64 <= K) __builtin_prefetch(brow + k + 32, 0, 3);
            acc = __builtin_amdgcn_wmma_f32_16x16x32_bf16(
                false, av, false, bv, (short)0, acc, false, false);
        }

        // C/D layout: VGPR v, lanes 0-15 -> M = v, lanes 16-31 -> M = v + 8; N = lane%16
        const int col = nt * 16 + r;
#pragma unroll
        for (int v = 0; v < 8; ++v) {
            const int rowi = mt * 16 + v + 8 * half;
            const float val = acc[v];
            if (EPI == 0) {
                outB[(size_t)rowi * N + col] = (bf16)val;
            } else if (EPI == 1) {
                outF[(size_t)rowi * N + col] = val;
                if (col < DT_RANK) aux[(size_t)rowi * DT_RANK + col] = (bf16)val;
            } else if (EPI == 2) {
                float t = val + bias[col];
                outF[(size_t)rowi * N + col] = (t > 20.f) ? t : log1pf(__expf(t));
            } else {
                outF[(size_t)rowi * N + col] = val + resid[(size_t)rowi * N + col];
            }
        }
    }
}

// ---------------- causal depthwise conv1d + SiLU ----------------
__global__ void conv_silu_kernel(const bf16* __restrict__ xz, const float* __restrict__ cw,
                                 const float* __restrict__ cb, bf16* __restrict__ u) {
    int gid = blockIdx.x * blockDim.x + threadIdx.x;   // ROWS * D_INNER threads
    int d    = gid & (D_INNER - 1);
    int row  = gid >> 11;                              // b*SEQ + t
    int t    = row & (SEQ - 1);
    int rb   = row - t;                                // b*SEQ
    float acc = cb[d];
#pragma unroll
    for (int j = 0; j < D_CONV; ++j) {
        int tt = t - (D_CONV - 1) + j;
        if (tt >= 0)
            acc += cw[d * D_CONV + j] * (float)xz[(size_t)(rb + tt) * (2 * D_INNER) + d];
    }
    float s = acc / (1.f + __expf(-acc));              // SiLU
    u[(size_t)row * D_INNER + d] = (bf16)s;
}

// ---------------- selective scan + skip + gate (one thread per (b,d)) ----------
__global__ void scan_kernel(const float* __restrict__ delta, const bf16* __restrict__ u,
                            const float* __restrict__ proj, const bf16* __restrict__ xz,
                            const float* __restrict__ A_log, const float* __restrict__ Dp,
                            bf16* __restrict__ yg) {
    int gid = blockIdx.x * blockDim.x + threadIdx.x;   // 0 .. B_SZ*D_INNER-1
    int d = gid & (D_INNER - 1);
    int b = gid >> 11;
    float Ad[D_STATE];
#pragma unroll
    for (int n = 0; n < D_STATE; ++n) Ad[n] = -__expf(A_log[d * D_STATE + n]);
    float h[D_STATE];
#pragma unroll
    for (int n = 0; n < D_STATE; ++n) h[n] = 0.f;
    const float Dd = Dp[d];

    for (int t = 0; t < SEQ; ++t) {
        const size_t row = (size_t)b * SEQ + t;
        const float dt = delta[row * D_INNER + d];
        const float uu = (float)u[row * D_INNER + d];
        const float zz = (float)xz[row * (2 * D_INNER) + D_INNER + d];
        const float* pb = proj + row * NPROJ + DT_RANK;        // B_t (broadcast in wave)
        if (t + 1 < SEQ) {
            __builtin_prefetch(proj + (row + 1) * NPROJ + DT_RANK, 0, 3);
            __builtin_prefetch(delta + (row + 1) * D_INNER + d, 0, 3);
        }
        const float du = dt * uu;
        float y = 0.f;
#pragma unroll
        for (int n = 0; n < D_STATE; ++n) {
            float hn = __expf(dt * Ad[n]) * h[n] + du * pb[n];
            h[n] = hn;
            y += hn * pb[D_STATE + n];                         // C_t
        }
        const float yv = y + uu * Dd;
        const float gate = zz / (1.f + __expf(-zz));           // SiLU(z)
        yg[row * D_INNER + d] = (bf16)(yv * gate);
    }
}

// ---------------------------------------------------------------------------
extern "C" void kernel_launch(void* const* d_in, const int* in_sizes, int n_in,
                              void* d_out, int out_size, void* d_ws, size_t ws_size,
                              hipStream_t stream) {
    const float* x       = (const float*)d_in[0];
    const float* ln_g    = (const float*)d_in[1];
    const float* ln_b    = (const float*)d_in[2];
    const float* W_in    = (const float*)d_in[3];
    const float* conv_w  = (const float*)d_in[4];
    const float* conv_b  = (const float*)d_in[5];
    const float* W_xproj = (const float*)d_in[6];
    const float* W_dt    = (const float*)d_in[7];
    const float* b_dt    = (const float*)d_in[8];
    const float* A_log   = (const float*)d_in[9];
    const float* D_param = (const float*)d_in[10];
    const float* W_out   = (const float*)d_in[11];
    float* out = (float*)d_out;

    char* ws = (char*)d_ws;
    bf16*  w_in_bf   = (bf16*)(ws + OFF_WIN);
    bf16*  w_xp_bf   = (bf16*)(ws + OFF_WXP);
    bf16*  w_dt_bf   = (bf16*)(ws + OFF_WDT);
    bf16*  w_out_bf  = (bf16*)(ws + OFF_WOUT);
    bf16*  xn_bf     = (bf16*)(ws + OFF_XN);
    bf16*  xz_bf     = (bf16*)(ws + OFF_XZ);
    bf16*  u_bf      = (bf16*)(ws + OFF_U);
    float* proj_f    = (float*)(ws + OFF_PROJ);
    bf16*  dtlow_bf  = (bf16*)(ws + OFF_DTLOW);
    float* delta_f   = (float*)(ws + OFF_DELTA);
    bf16*  yg_bf     = (bf16*)(ws + OFF_YG);

    // 1) weights -> bf16
    {
        int n1 = 2 * D_INNER * D_MODEL, n2 = NPROJ * D_INNER,
            n3 = D_INNER * DT_RANK,     n4 = D_MODEL * D_INNER;
        f2bf_kernel<<<(n1 + 255) / 256, 256, 0, stream>>>(W_in,    w_in_bf,  n1);
        f2bf_kernel<<<(n2 + 255) / 256, 256, 0, stream>>>(W_xproj, w_xp_bf,  n2);
        f2bf_kernel<<<(n3 + 255) / 256, 256, 0, stream>>>(W_dt,    w_dt_bf,  n3);
        f2bf_kernel<<<(n4 + 255) / 256, 256, 0, stream>>>(W_out,   w_out_bf, n4);
    }

    // 2) LayerNorm
    ln_kernel<<<ROWS, 256, 0, stream>>>(x, ln_g, ln_b, xn_bf);

    // 3) GEMM1: xz = xn @ W_in^T  (4096x1024 * 1024x4096)
    {
        dim3 grid(((2 * D_INNER) / 16 + 7) / 8, ROWS / 16);
        wmma_gemm_tdm_kernel<0><<<grid, 256, 16 * D_MODEL * 2, stream>>>(
            xn_bf, w_in_bf, 2 * D_INNER, D_MODEL,
            nullptr, nullptr, nullptr, xz_bf, nullptr);
    }

    // 4) conv + SiLU -> u
    conv_silu_kernel<<<(ROWS * D_INNER) / 256, 256, 0, stream>>>(xz_bf, conv_w, conv_b, u_bf);

    // 5) GEMM2: proj = u @ W_xproj^T  (4096x2048 * 2048x160); dt_low mirrored as bf16
    {
        dim3 grid((NPROJ / 16 + 7) / 8, ROWS / 16);
        wmma_gemm_tdm_kernel<1><<<grid, 256, 16 * D_INNER * 2, stream>>>(
            u_bf, w_xp_bf, NPROJ, D_INNER,
            nullptr, nullptr, proj_f, nullptr, dtlow_bf);
    }

    // 6) GEMM3: delta = softplus(dt_low @ W_dt^T + b_dt)  (4096x128 * 128x2048)
    {
        dim3 grid((D_INNER / 16 + 7) / 8, ROWS / 16);
        wmma_gemm_tdm_kernel<2><<<grid, 256, 16 * DT_RANK * 2, stream>>>(
            dtlow_bf, w_dt_bf, D_INNER, DT_RANK,
            b_dt, nullptr, delta_f, nullptr, nullptr);
    }

    // 7) selective scan + u*D + SiLU(z) gate -> yg (bf16)
    scan_kernel<<<(B_SZ * D_INNER) / 256, 256, 0, stream>>>(
        delta_f, u_bf, proj_f, xz_bf, A_log, D_param, yg_bf);

    // 8) GEMM4: out = resid + yg @ W_out^T  (4096x2048 * 2048x1024)
    {
        dim3 grid((D_MODEL / 16 + 7) / 8, ROWS / 16);
        wmma_gemm_tdm_kernel<3><<<grid, 256, 16 * D_INNER * 2, stream>>>(
            yg_bf, w_out_bf, D_MODEL, D_INNER,
            nullptr, x, out, nullptr, nullptr);
    }

    (void)in_sizes; (void)n_in; (void)out_size; (void)ws_size;
}
